// Attention_609885356587
// MI455X (gfx1250) — matrix operand order repaired and checked by reading, not compiled, not run on hardware
//
#include <hip/hip_runtime.h>
#include <hip/hip_bf16.h>
#include <stdint.h>

typedef __attribute__((ext_vector_type(16))) __bf16 v16bf;
typedef __attribute__((ext_vector_type(8)))  __bf16 v8bf;
typedef __attribute__((ext_vector_type(8)))  float  v8f;

#define HEADS  8
#define DHEAD  64
#define DIM    512
#define NSEQ   4096
#define BATCH  2
#define NROWS  (BATCH * NSEQ)   /* 8192 */
#define QKVN   (3 * DIM)        /* 1536 */
#define QSCALE 0.044194173824159216f /* 512^-0.5 */

__device__ __forceinline__ unsigned short f32_to_bf16(float f) {
    union { __bf16 h; unsigned short u; } c;
    c.h = (__bf16)f;
    return c.u;
}
// One v_cvt_pk_bf16_f32, no repacking
__device__ __forceinline__ uint32_t pack_bf16x2(float a, float b) {
#if __has_builtin(__builtin_amdgcn_cvt_pk_bf16_f32)
    union { decltype(__builtin_amdgcn_cvt_pk_bf16_f32(0.f, 0.f)) v; uint32_t u; } c;
    c.v = __builtin_amdgcn_cvt_pk_bf16_f32(a, b);
    return c.u;
#else
    union { __bf16 h[2]; uint32_t u; } c;
    c.h[0] = (__bf16)a; c.h[1] = (__bf16)b;
    return c.u;
#endif
}
__device__ __forceinline__ v16bf concat8(v8bf lo, v8bf hi) {
    return __builtin_shufflevector(lo, hi, 0,1,2,3,4,5,6,7,8,9,10,11,12,13,14,15);
}
// Two contiguous 16B runs -> one v16bf fragment (2x ds_load_b128)
__device__ __forceinline__ v16bf load_frag2(const unsigned short* p0,
                                            const unsigned short* p1) {
    return concat8(*(const v8bf*)p0, *(const v8bf*)p1);
}

// ---------------------------------------------------------------------------
// Kernel 1: qkv = x @ Wqkv  (f32 in, bf16 out, head-major, Q pre-scaled)
// 128x128 tile, 256 threads = 8 waves; peeled software pipeline (no phis).
// ---------------------------------------------------------------------------
__global__ __launch_bounds__(256)
void qkv_gemm_kernel(const float* __restrict__ x, const float* __restrict__ Wqkv,
                     unsigned short* __restrict__ Qws,
                     unsigned short* __restrict__ Kws,
                     unsigned short* __restrict__ Vws) {
    // pitch 40 ushorts = 80 B (multiple of 16B) keeps b128 loads aligned
    __shared__ __attribute__((aligned(16))) unsigned short As[128][40];   // [m][k]
    __shared__ __attribute__((aligned(16))) unsigned short Bt[128][40];   // [n][k]

    const int tid  = threadIdx.x;
    const int wid  = tid >> 5;
    const int lane = tid & 31;
    const int half = lane >> 4;
    const int l16  = lane & 15;
    const int mtile = blockIdx.x % 64;           // 8192/128
    const int ntile = blockIdx.x / 64;           // 1536/128
    const int m0 = mtile * 128, n0 = ntile * 128;

    v8f acc[8];
    #pragma unroll
    for (int i = 0; i < 8; ++i)
        #pragma unroll
        for (int j = 0; j < 8; ++j) acc[i][j] = 0.0f;

    float  ax[16];
    float2 blo[4], bhi[4];
    auto load_tile = [&](int k0) {
        #pragma unroll
        for (int i = 0; i < 8; ++i) {            // A: 128x32 f32, float2 along k
            int e = tid + i * 256;
            int r = e >> 4, c2 = (e & 15) * 2;
            float2 v = *(const float2*)&x[(size_t)(m0 + r) * DIM + k0 + c2];
            ax[2 * i] = v.x; ax[2 * i + 1] = v.y;
        }
        #pragma unroll
        for (int i = 0; i < 4; ++i) {            // B: 2x2 unit (2 rows x 2 cols)
            int e = tid + i * 256;               // 1024 units
            int c2 = (e & 63) * 2, r2 = (e >> 6) * 2;
            const float* wp = &Wqkv[(size_t)(k0 + r2) * QKVN + n0 + c2];
            blo[i] = *(const float2*)wp;
            bhi[i] = *(const float2*)(wp + QKVN);
        }
    };
    auto store_tile = [&]() {
        #pragma unroll
        for (int i = 0; i < 8; ++i) {
            int e = tid + i * 256;
            int r = e >> 4, c2 = (e & 15) * 2;
            *(uint32_t*)&As[r][c2] = pack_bf16x2(ax[2 * i], ax[2 * i + 1]);
        }
        #pragma unroll
        for (int i = 0; i < 4; ++i) {            // Bt[n][k]: one b32 per column
            int e = tid + i * 256;
            int c2 = (e & 63) * 2, r2 = (e >> 6) * 2;
            *(uint32_t*)&Bt[c2][r2]     = pack_bf16x2(blo[i].x, bhi[i].x);
            *(uint32_t*)&Bt[c2 + 1][r2] = pack_bf16x2(blo[i].y, bhi[i].y);
        }
    };
    auto compute = [&]() {
        const unsigned short* arow = &As[wid * 16 + l16][0];
        v16bf a = load_frag2(arow + half * 8, arow + 16 + half * 8);
        #pragma unroll
        for (int ns = 0; ns < 8; ++ns) {
            const unsigned short* brow = &Bt[ns * 16 + l16][half * 8];
            v16bf b = load_frag2(brow, brow + 16);
            acc[ns] = __builtin_amdgcn_wmma_f32_16x16x32_bf16(
                false, a, false, b, (short)0, acc[ns], false, false);
        }
    };

    load_tile(0);
    store_tile();
    for (int k0 = 0; k0 < DIM - 32; k0 += 32) {  // peeled pipeline
        __syncthreads();
        load_tile(k0 + 32);
        compute();
        __syncthreads();
        store_tile();
    }
    __syncthreads();
    compute();                                   // final tile

    // Scatter D fragments into Q/K/V [b][h][n][d] bf16 (uniform select per ns)
    #pragma unroll
    for (int ns = 0; ns < 8; ++ns) {
        int colbase = n0 + ns * 16;              // 16-wide subtile, one buffer
        int sel = colbase >> 9;
        unsigned short* dstbuf = (sel == 0) ? Qws : (sel == 1) ? Kws : Vws;
        float scl = (sel == 0) ? QSCALE : 1.0f;
        int cc = colbase & 511;
        int hh = cc >> 6, dd = (cc & 63) + l16;
        #pragma unroll
        for (int r = 0; r < 8; ++r) {
            int row = m0 + wid * 16 + r + half * 8;   // D: M = r + 8*half
            int bb = row >> 12, nn = row & 4095;
            size_t dst = (((size_t)bb * HEADS + hh) * NSEQ + nn) * DHEAD + dd;
            dstbuf[dst] = f32_to_bf16(acc[ns][r] * scl);
        }
    }
}

// ---------------------------------------------------------------------------
// Kernel 2: flash attention. 1 block per (b,h,128-query tile).
// All queries attend to k_t (keys 2048..4095); values from v_s or v_t.
// Each wave: 16 queries x 64 dims. Peeled K/V software pipeline.
// ---------------------------------------------------------------------------
__global__ __launch_bounds__(256)
void attn_kernel(const unsigned short* __restrict__ Qws,
                 const unsigned short* __restrict__ Kws,
                 const unsigned short* __restrict__ Vws,
                 unsigned short* __restrict__ AO) {
    __shared__ __attribute__((aligned(16))) unsigned short Kblk[32][72]; // [key][dim]
    __shared__ __attribute__((aligned(16))) unsigned short Vt[64][40];   // [dim][key]
    __shared__ __attribute__((aligned(16))) unsigned short Pst[8][16][40];

    const int tid  = threadIdx.x;
    const int wid  = tid >> 5;
    const int lane = tid & 31;
    const int half = lane >> 4;
    const int l16  = lane & 15;

    const int qt = blockIdx.x & 31;          // 4096/128 query tiles
    const int bh = blockIdx.x >> 5;          // 0..15
    const int bb = bh >> 3, hh = bh & 7;
    const int qbase = qt * 128;

    const size_t headoff = ((size_t)bb * HEADS + hh) * (size_t)NSEQ * DHEAD;
    const unsigned short* Qh = Qws + headoff;
    const unsigned short* Kh = Kws + headoff + (size_t)2048 * DHEAD;  // k_t
    const unsigned short* Vh = Vws + headoff +
                               ((qbase < 2048) ? (size_t)0 : (size_t)2048 * DHEAD);

    // Q A-fragments (2 chunks over the 64-dim contraction): 2x global b128 each
    v16bf qf[2];
    {
        const int qrow = qbase + wid * 16 + l16;
        const unsigned short* qp = Qh + (size_t)qrow * DHEAD;
        #pragma unroll
        for (int kc = 0; kc < 2; ++kc)
            qf[kc] = load_frag2(qp + kc * 32 + half * 8,
                                qp + kc * 32 + 16 + half * 8);
    }

    v8f acc[4];
    float mrow[8], lrow[8];
    #pragma unroll
    for (int i = 0; i < 4; ++i)
        #pragma unroll
        for (int j = 0; j < 8; ++j) acc[i][j] = 0.0f;
    #pragma unroll
    for (int r = 0; r < 8; ++r) { mrow[r] = -1.0e30f; lrow[r] = 0.0f; }

    uint32_t kReg[4], vLo[2], vHi[2];
    auto load_kv = [&](int j0) {
        #pragma unroll
        for (int i = 0; i < 4; ++i) {            // K row-major: dword copies
            int e = tid + i * 256;
            int r = e >> 5, c2 = (e & 31) * 2;
            kReg[i] = *(const uint32_t*)&Kh[(size_t)(j0 + r) * DHEAD + c2];
        }
        #pragma unroll
        for (int i = 0; i < 2; ++i) {            // V: 2x2 unit (2 keys x 2 dims)
            int e = tid + i * 256;               // 512 units
            int c2 = (e & 31) * 2, r2 = (e >> 5) * 2;
            const unsigned short* vp = &Vh[(size_t)(j0 + r2) * DHEAD + c2];
            vLo[i] = *(const uint32_t*)vp;
            vHi[i] = *(const uint32_t*)(vp + DHEAD);
        }
    };
    auto store_kv = [&]() {
        #pragma unroll
        for (int i = 0; i < 4; ++i) {
            int e = tid + i * 256;
            int r = e >> 5, c2 = (e & 31) * 2;
            *(uint32_t*)&Kblk[r][c2] = kReg[i];
        }
        #pragma unroll
        for (int i = 0; i < 2; ++i) {            // transpose via v_perm_b32
            int e = tid + i * 256;
            int c2 = (e & 31) * 2, r2 = (e >> 5) * 2;
            *(uint32_t*)&Vt[c2][r2] =
                __builtin_amdgcn_perm(vHi[i], vLo[i], 0x05040100u);
            *(uint32_t*)&Vt[c2 + 1][r2] =
                __builtin_amdgcn_perm(vHi[i], vLo[i], 0x07060302u);
        }
    };

    auto process_block = [&]() {
        // S = Q * K^T for 32 keys (2 x 16-key subtiles, 2 dim-chunks each)
        v8f s[2];
        #pragma unroll
        for (int ns = 0; ns < 2; ++ns) {
            #pragma unroll
            for (int j = 0; j < 8; ++j) s[ns][j] = 0.0f;
            #pragma unroll
            for (int kc = 0; kc < 2; ++kc) {
                const unsigned short* krow = &Kblk[ns * 16 + l16][kc * 32 + half * 8];
                v16bf b = load_frag2(krow, krow + 16);  // B[d][key]
                s[ns] = __builtin_amdgcn_wmma_f32_16x16x32_bf16(
                    false, qf[kc], false, b, (short)0, s[ns], false, false);
            }
        }

        // Online softmax: row stats reduced across the 16-lane half groups
        float bm[8];
        #pragma unroll
        for (int r = 0; r < 8; ++r) bm[r] = fmaxf(s[0][r], s[1][r]);
        #pragma unroll
        for (int off = 1; off < 16; off <<= 1)
            #pragma unroll
            for (int r = 0; r < 8; ++r)
                bm[r] = fmaxf(bm[r], __shfl_xor(bm[r], off, 32));

        float corr[8];
        #pragma unroll
        for (int r = 0; r < 8; ++r) {
            float mn = fmaxf(mrow[r], bm[r]);
            corr[r] = __expf(mrow[r] - mn);
            mrow[r] = mn;
        }
        #pragma unroll
        for (int ns = 0; ns < 2; ++ns)
            #pragma unroll
            for (int r = 0; r < 8; ++r)
                s[ns][r] = __expf(s[ns][r] - mrow[r]);

        float rs[8];
        #pragma unroll
        for (int r = 0; r < 8; ++r) rs[r] = s[0][r] + s[1][r];
        #pragma unroll
        for (int off = 1; off < 16; off <<= 1)
            #pragma unroll
            for (int r = 0; r < 8; ++r)
                rs[r] += __shfl_xor(rs[r], off, 32);
        #pragma unroll
        for (int r = 0; r < 8; ++r) lrow[r] = lrow[r] * corr[r] + rs[r];
        #pragma unroll
        for (int i = 0; i < 4; ++i)
            #pragma unroll
            for (int r = 0; r < 8; ++r) acc[i][r] *= corr[r];

        // Restage P (16x32) through wave-private LDS (in-order per wave;
        // no block barrier needed) to re-form an A fragment
        #pragma unroll
        for (int ns = 0; ns < 2; ++ns)
            #pragma unroll
            for (int r = 0; r < 8; ++r)
                Pst[wid][r + half * 8][ns * 16 + l16] = f32_to_bf16(s[ns][r]);

        const unsigned short* prow = &Pst[wid][l16][0];
        v16bf pf = load_frag2(prow + half * 8, prow + 16 + half * 8);
        #pragma unroll
        for (int ns2 = 0; ns2 < 4; ++ns2) {
            const unsigned short* vrow = &Vt[ns2 * 16 + l16][half * 8];
            v16bf b = load_frag2(vrow, vrow + 16);      // B[key][dim] via Vt
            acc[ns2] = __builtin_amdgcn_wmma_f32_16x16x32_bf16(
                false, pf, false, b, (short)0, acc[ns2], false, false);
        }
    };

    load_kv(0);
    store_kv();
    for (int j0 = 0; j0 < 2048 - 32; j0 += 32) { // peeled pipeline
        __syncthreads();
        load_kv(j0 + 32);
        process_block();
        __syncthreads();
        store_kv();
    }
    __syncthreads();
    process_block();                             // final key block

    // Normalize and store AO in [b*n][h*d] bf16 for the projection GEMM
    #pragma unroll
    for (int ns2 = 0; ns2 < 4; ++ns2)
        #pragma unroll
        for (int r = 0; r < 8; ++r) {
            float o = acc[ns2][r] / lrow[r];
            int row = qbase + wid * 16 + r + half * 8;
            size_t dst = ((size_t)bb * NSEQ + row) * DIM + hh * DHEAD + ns2 * 16 + l16;
            AO[dst] = f32_to_bf16(o);
        }
}

// ---------------------------------------------------------------------------
// Kernel 3: out = AO @ Wout + b_out  (bf16 x bf16 -> f32), pipelined like k1
// ---------------------------------------------------------------------------
__global__ __launch_bounds__(256)
void out_gemm_kernel(const unsigned short* __restrict__ AO,
                     const float* __restrict__ Wout,
                     const float* __restrict__ bout,
                     float* __restrict__ out) {
    __shared__ __attribute__((aligned(16))) unsigned short As[128][40];
    __shared__ __attribute__((aligned(16))) unsigned short Bt[128][40];

    const int tid  = threadIdx.x;
    const int wid  = tid >> 5;
    const int lane = tid & 31;
    const int half = lane >> 4;
    const int l16  = lane & 15;
    const int mtile = blockIdx.x % 64;           // 8192/128
    const int ntile = blockIdx.x / 64;           // 512/128
    const int m0 = mtile * 128, n0 = ntile * 128;

    v8f acc[8];
    #pragma unroll
    for (int i = 0; i < 8; ++i)
        #pragma unroll
        for (int j = 0; j < 8; ++j) acc[i][j] = 0.0f;

    uint32_t aU[8];
    float2 blo[4], bhi[4];
    auto load_tile = [&](int k0) {
        #pragma unroll
        for (int i = 0; i < 8; ++i) {            // A already bf16: dword copies
            int e = tid + i * 256;
            int r = e >> 4, c2 = (e & 15) * 2;
            aU[i] = *(const uint32_t*)&AO[(size_t)(m0 + r) * DIM + k0 + c2];
        }
        #pragma unroll
        for (int i = 0; i < 4; ++i) {            // B: 2x2 unit
            int e = tid + i * 256;
            int c2 = (e & 63) * 2, r2 = (e >> 6) * 2;
            const float* wp = &Wout[(size_t)(k0 + r2) * DIM + n0 + c2];
            blo[i] = *(const float2*)wp;
            bhi[i] = *(const float2*)(wp + DIM);
        }
    };
    auto store_tile = [&]() {
        #pragma unroll
        for (int i = 0; i < 8; ++i) {
            int e = tid + i * 256;
            int r = e >> 4, c2 = (e & 15) * 2;
            *(uint32_t*)&As[r][c2] = aU[i];
        }
        #pragma unroll
        for (int i = 0; i < 4; ++i) {
            int e = tid + i * 256;
            int c2 = (e & 63) * 2, r2 = (e >> 6) * 2;
            *(uint32_t*)&Bt[c2][r2]     = pack_bf16x2(blo[i].x, bhi[i].x);
            *(uint32_t*)&Bt[c2 + 1][r2] = pack_bf16x2(blo[i].y, bhi[i].y);
        }
    };
    auto compute = [&]() {
        const unsigned short* arow = &As[wid * 16 + l16][0];
        v16bf a = load_frag2(arow + half * 8, arow + 16 + half * 8);
        #pragma unroll
        for (int ns = 0; ns < 8; ++ns) {
            const unsigned short* brow = &Bt[ns * 16 + l16][half * 8];
            v16bf b = load_frag2(brow, brow + 16);
            acc[ns] = __builtin_amdgcn_wmma_f32_16x16x32_bf16(
                false, a, false, b, (short)0, acc[ns], false, false);
        }
    };

    load_tile(0);
    store_tile();
    for (int k0 = 0; k0 < DIM - 32; k0 += 32) {
        __syncthreads();
        load_tile(k0 + 32);
        compute();
        __syncthreads();
        store_tile();
    }
    __syncthreads();
    compute();

    #pragma unroll
    for (int ns = 0; ns < 8; ++ns)
        #pragma unroll
        for (int r = 0; r < 8; ++r) {
            int row = m0 + wid * 16 + r + half * 8;
            int col = n0 + ns * 16 + l16;
            out[(size_t)row * DIM + col] = acc[ns][r] + bout[col];
        }
}

// ---------------------------------------------------------------------------
extern "C" void kernel_launch(void* const* d_in, const int* in_sizes, int n_in,
                              void* d_out, int out_size, void* d_ws, size_t ws_size,
                              hipStream_t stream) {
    (void)in_sizes; (void)n_in; (void)out_size; (void)ws_size;
    const float* x     = (const float*)d_in[0];
    const float* Wqkv  = (const float*)d_in[1];
    const float* Wout  = (const float*)d_in[2];
    const float* b_out = (const float*)d_in[3];
    float* out = (float*)d_out;

    const size_t TEN = (size_t)BATCH * HEADS * NSEQ * DHEAD;  // 4,194,304 elems
    unsigned short* Qws = (unsigned short*)d_ws;
    unsigned short* Kws = Qws + TEN;
    unsigned short* Vws = Kws + TEN;
    unsigned short* AO  = Vws + TEN;                          // 8192x512

    hipLaunchKernelGGL(qkv_gemm_kernel, dim3(64 * 12), dim3(256), 0, stream,
                       x, Wqkv, Qws, Kws, Vws);
    hipLaunchKernelGGL(attn_kernel, dim3(32 * 16), dim3(256), 0, stream,
                       Qws, Kws, Vws, AO);
    hipLaunchKernelGGL(out_gemm_kernel, dim3(64 * 4), dim3(256), 0, stream,
                       AO, Wout, b_out, out);
}
